// GatingConv_32547262169679
// MI455X (gfx1250) — matrix operand order repaired and checked by reading, not compiled, not run on hardware
//
#include <hip/hip_runtime.h>
#include <hip/hip_bf16.h>

typedef __attribute__((ext_vector_type(16))) _Float16 v16h;
typedef __attribute__((ext_vector_type(8)))  float    v8f;

#define EPSV 1e-5f

__device__ __forceinline__ float gelu_erf(float x) {
    return 0.5f * x * (1.0f + erff(x * 0.70710678118654752f));
}

// ---------------------------------------------------------------------------
// weight repack f32 [Cout][Ktot] -> f16 [Cout][Kpad] (zero-padded K)
// ---------------------------------------------------------------------------
__global__ void cvt_w_kernel(const float* __restrict__ src, _Float16* __restrict__ dst,
                             int Cout, int Ktot, int Kpad) {
    int idx = blockIdx.x * blockDim.x + threadIdx.x;
    if (idx >= Cout * Kpad) return;
    int k = idx % Kpad; int co = idx / Kpad;
    dst[idx] = (k < Ktot) ? (_Float16)src[co * Ktot + k] : (_Float16)0.0f;
}

// ---------------------------------------------------------------------------
// Implicit-GEMM conv via V_WMMA_F32_16X16X32_F16, LDS-staged B tiles.
//
//   A (16xK)  = f16 weights, row = out-channel (M);    global b32 loads
//   B (Kx16)  = f16 im2col patches, col = pixel (N);   cooperative gather ->
//               LDS in fragment order [n][k], double buffered, 1 barrier/step
//   D (16x16) = f32 accum; fused BN + erf-GELU epilogue.
//
// Block = 256 threads = 8 waves = MTW m-tiles x (NT/2) wave-columns; each wave
// owns one 16-row M tile and TWO 16-col N tiles (acc0/acc1) sharing its A
// fragment -> 2 back-to-back v_wmma per K-step.
//
// Fragment layouts used here (wave32, K contiguous per lane half):
//   A: lane half h, VGPR i: K = 16h + 2*i + {0,1}
//   B: lane half h, elem j:  K = 16h + j,  N = lane&15
//   C/D: VGPR r: M = 8h + r, N = lane&15
//
// All uses have Cout%16==0, (64*OH*OW) % (16*NT) == 0 -> no guards, no
// divergent barriers. EXEC is all-ones at every WMMA.
// ---------------------------------------------------------------------------
template<typename TIN, int CIN, int IH, int IW, int COUT, int OH, int OW,
         int SH, int SW, int PH, int PW, int KHT, int KWT,
         int KTOT, int KPAD, int MTW, int NT, int MBLK, int ACT>
__global__ __launch_bounds__(256)
void conv_wmma_kernel(const TIN* __restrict__ in, const _Float16* __restrict__ wgt,
                      const float* __restrict__ gma, const float* __restrict__ bta,
                      _Float16* __restrict__ outH, float* __restrict__ outF)
{
    constexpr int KHW     = KHT * KWT;
    constexpr int KSTEPS  = KPAD / 32;
    constexpr int E       = NT * 2;            // gathered elems / thread / step
    constexpr int BUFELEM = NT * 16 * 32;
    __shared__ _Float16 Bt[2][NT][16][32];     // [buf][ni][n][k]

    const int tid  = threadIdx.x;
    const int lane = tid & 31;
    const int w    = tid >> 5;
    const int half = lane >> 4;
    const int lid  = lane & 15;

    const int mBlk   = (int)blockIdx.x % MBLK;
    const int nBlk   = (int)blockIdx.x / MBLK;
    const int tileM  = mBlk * MTW + (w % MTW);
    const int nPair  = w / MTW;
    const int tileN0 = nBlk * NT + nPair * 2;  // this wave's two N tiles

    // ---- per-thread gather descriptors (K-invariant, hoisted) ----
    int        kLoc[E];
    int        ih0a[E], iw0a[E];
    const TIN* basep[E];
    _Float16*  ldsp[E];
    #pragma unroll
    for (int i = 0; i < E; ++i) {
        int eid  = i * 256 + tid;
        int kl   = eid & 31;
        int rest = eid >> 5;
        int nLoc = rest & 15;
        int ni   = rest >> 4;
        int n    = (nBlk * NT + ni) * 16 + nLoc;
        int ow   = n % OW;                     // constant divisors
        int oh   = (n / OW) % OH;
        int bb   = n / (OW * OH);
        kLoc[i]  = kl;
        ih0a[i]  = oh * SH - PH;
        iw0a[i]  = ow * SW - PW;
        basep[i] = in + (size_t)bb * CIN * IH * IW;
        ldsp[i]  = &Bt[0][ni][nLoc][kl];
    }

    auto gather = [&](int ks, int buf) {
        const int kbase = ks * 32;
        #pragma unroll
        for (int i = 0; i < E; ++i) {
            int kk = kbase + kLoc[i];
            _Float16 v = (_Float16)0.0f;
            if (KTOT == KPAD || kk < KTOT) {   // constexpr-folds when no K pad
                int c  = kk / KHW;
                int rs = kk - c * KHW;
                int r  = rs / KWT;
                int s  = rs - r * KWT;
                int ih = ih0a[i] + r, iw = iw0a[i] + s;
                if ((unsigned)ih < (unsigned)IH && (unsigned)iw < (unsigned)IW)
                    v = (_Float16)basep[i][(c * IH + ih) * IW + iw];
            }
            ldsp[i][buf * BUFELEM] = v;
        }
    };

    const _Float16* wrow  = wgt + (size_t)(tileM * 16 + lid) * KPAD;
    const _Float16* bptr0 = &Bt[0][nPair * 2    ][lid][16 * half];
    const _Float16* bptr1 = &Bt[0][nPair * 2 + 1][lid][16 * half];

    v8f acc0 = {}, acc1 = {};
    gather(0, 0);
    __syncthreads();
    for (int ks = 0; ks < KSTEPS; ++ks) {
        const int buf = ks & 1;
        if (ks + 1 < KSTEPS) gather(ks + 1, buf ^ 1);

        union { v16h v; unsigned int u[8]; } A;
        #pragma unroll
        for (int i = 0; i < 8; ++i) {
            int kk = ks * 32 + (half << 4) + (i << 1);
            A.u[i] = *(const unsigned int*)(wrow + kk);  // 2 consecutive K f16
        }
        v16h b0 = *(const v16h*)(bptr0 + buf * BUFELEM); // 32B contiguous LDS
        v16h b1 = *(const v16h*)(bptr1 + buf * BUFELEM);

        acc0 = __builtin_amdgcn_wmma_f32_16x16x32_f16(
            false, A.v, false, b0, (short)0, acc0, false, false);
        acc1 = __builtin_amdgcn_wmma_f32_16x16x32_f16(
            false, A.v, false, b1, (short)0, acc1, false, false);
        __syncthreads();   // readers done with buf before it is re-gathered
    }

    // ---- epilogue: BN + erf-GELU -> f16 (conv) | raw f32 (stem) ----
    auto store_tile = [&](v8f& acc, int tileN) {
        int n  = tileN * 16 + lid;
        int ow = n % OW, oh = (n / OW) % OH, bb = n / (OW * OH);
        size_t pix = (size_t)bb * COUT * OH * OW + (size_t)oh * OW + ow;
        #pragma unroll
        for (int r = 0; r < 8; ++r) {
            int mo  = tileM * 16 + (half << 3) + r;
            float v = acc[r];
            size_t o = pix + (size_t)mo * OH * OW;
            if (ACT) {
                float sc = gma[mo] * rsqrtf(1.0f + EPSV);
                outH[o] = (_Float16)gelu_erf(v * sc + bta[mo]);
            } else {
                outF[o] = v;
            }
        }
    };
    store_tile(acc0, tileN0);
    store_tile(acc1, tileN0 + 1);
}

// ---------------------------------------------------------------------------
// head: AdaptiveAvgPool(1,1) + FC(256->32)+GELU+BN1d + FC(32->5) + top-2.
// ---------------------------------------------------------------------------
__global__ void head_kernel(const _Float16* __restrict__ h4,
                            const float* __restrict__ fw1, const float* __restrict__ fb1,
                            const float* __restrict__ g1d, const float* __restrict__ b1d,
                            const float* __restrict__ fw2, const float* __restrict__ fb2,
                            float* __restrict__ rw_out, int* __restrict__ sel)
{
    __shared__ float pooled[256];
    __shared__ float hid[32];
    __shared__ float rw[5];
    const int b = blockIdx.x;
    const int t = threadIdx.x;

    {   // mean over 5x5 spatial (25 contiguous f16 per channel)
        const _Float16* p = h4 + ((size_t)b * 256 + t) * 25;
        float s = 0.f;
        for (int i = 0; i < 25; ++i) s += (float)p[i];
        pooled[t] = s * (1.0f / 25.0f);
    }
    __syncthreads();
    if (t < 32) {
        float s = fb1[t];
        for (int c = 0; c < 256; ++c) s += pooled[c] * fw1[t * 256 + c];
        s = gelu_erf(s);
        hid[t] = s * (g1d[t] * rsqrtf(1.0f + EPSV)) + b1d[t];
    }
    __syncthreads();
    if (t < 5) {
        float s = fb2[t];
        for (int i = 0; i < 32; ++i) s += hid[i] * fw2[t * 32 + i];
        rw[t] = s;
        rw_out[b * 5 + t] = s;
    }
    __syncthreads();
    if (t == 0) {
        // top-2 of softmax == top-2 of logits; lower index wins ties
        int i1 = 0;
        for (int i = 1; i < 5; ++i) if (rw[i] > rw[i1]) i1 = i;
        int i2 = (i1 == 0) ? 1 : 0;
        for (int i = 0; i < 5; ++i) if (i != i1 && rw[i] > rw[i2]) i2 = i;
        sel[b * 2 + 0] = min(i1, i2);
        sel[b * 2 + 1] = max(i1, i2);
    }
}

// ---------------------------------------------------------------------------
// gather: gated[b,f, t*2+s] = x[b,0,f,t, sel[b][s]]  (f32 -> f16)
// ---------------------------------------------------------------------------
__global__ void gather_kernel(const float* __restrict__ x, const int* __restrict__ sel,
                              _Float16* __restrict__ gated) {
    const int W = 800, F = 80;
    int idx = blockIdx.x * blockDim.x + threadIdx.x;
    if (idx >= 64 * F * W) return;
    int w = idx % W; int t2 = idx / W;
    int f = t2 % F;  int b  = t2 / F;
    int tt = w >> 1, s = w & 1;
    int nsel = sel[b * 2 + s];
    gated[idx] = (_Float16)x[(((size_t)b * F + f) * 400 + tt) * 5 + nsel];
}

// ---------------------------------------------------------------------------
extern "C" void kernel_launch(void* const* d_in, const int* in_sizes, int n_in,
                              void* d_out, int out_size, void* d_ws, size_t ws_size,
                              hipStream_t stream) {
    (void)in_sizes; (void)n_in; (void)out_size; (void)ws_size;
    const float* x     = (const float*)d_in[0];
    const float* total = (const float*)d_in[1];
    const float* gw1 = (const float*)d_in[2];
    const float* g1  = (const float*)d_in[3];  const float* b1  = (const float*)d_in[4];
    const float* gw2 = (const float*)d_in[5];
    const float* g2  = (const float*)d_in[6];  const float* b2  = (const float*)d_in[7];
    const float* gw3 = (const float*)d_in[8];
    const float* g3  = (const float*)d_in[9];  const float* b3  = (const float*)d_in[10];
    const float* gw4 = (const float*)d_in[11];
    const float* g4  = (const float*)d_in[12]; const float* b4  = (const float*)d_in[13];
    const float* fw1 = (const float*)d_in[14]; const float* fb1 = (const float*)d_in[15];
    const float* g1d = (const float*)d_in[16]; const float* b1d = (const float*)d_in[17];
    const float* fw2 = (const float*)d_in[18]; const float* fb2 = (const float*)d_in[19];
    const float* cw1 = (const float*)d_in[20];

    float* out_conv = (float*)d_out;                       // (64,64,40,397)
    float* out_rw   = (float*)d_out + (size_t)65044480;    // (64,5)

    // ---- workspace layout (256B aligned), ~122 MB ----
    char* ws = (char*)d_ws;
    size_t off = 0;
    auto take = [&](size_t bytes) { size_t r = off; off += (bytes + 255) & ~(size_t)255; return r; };
    _Float16* h1  = (_Float16*)(ws + take((size_t)40796160 * 2));  // (64,32,40,498)
    _Float16* h2  = (_Float16*)(ws + take((size_t)9994240  * 2));  // (64,64,20,122)
    _Float16* h3  = (_Float16*)(ws + take((size_t)2293760  * 2));  // (64,128,10,28)
    _Float16* h4  = (_Float16*)(ws + take((size_t)409600   * 2));  // (64,256,5,5)
    _Float16* w1h = (_Float16*)(ws + take((size_t)(32*96)  * 2));  // 32 x 96  (K 75 pad)
    _Float16* w2h = (_Float16*)(ws + take((size_t)153600   * 2));  // 64 x 2400
    _Float16* w3h = (_Float16*)(ws + take((size_t)614400   * 2));  // 128 x 4800
    _Float16* w4h = (_Float16*)(ws + take((size_t)2457600  * 2));  // 256 x 9600
    _Float16* wch = (_Float16*)(ws + take((size_t)(64*128) * 2));  // 64 x 128 (K 98 pad)
    _Float16* gt  = (_Float16*)(ws + take((size_t)4096000  * 2));  // (64,1,80,800)
    int*      sel = (int*)     (ws + take((size_t)64 * 2 * 4));

    // ---- weight repacks ----
    cvt_w_kernel<<<(32 * 96    + 255) / 256, 256, 0, stream>>>(gw1, w1h, 32, 75, 96);
    cvt_w_kernel<<<(64 * 2400  + 255) / 256, 256, 0, stream>>>(gw2, w2h, 64, 2400, 2400);
    cvt_w_kernel<<<(128 * 4800 + 255) / 256, 256, 0, stream>>>(gw3, w3h, 128, 4800, 4800);
    cvt_w_kernel<<<(256 * 9600 + 255) / 256, 256, 0, stream>>>(gw4, w4h, 256, 9600, 9600);
    cvt_w_kernel<<<(64 * 128   + 255) / 256, 256, 0, stream>>>(cw1, wch, 64, 98, 128);

    // ---- conv1: total(f32) -> h1 ----  nTiles=79680, NT=8 -> 9960 blocks
    conv_wmma_kernel<float, 1, 80, 2000, 32, 40, 498, 2, 4, 2, 2, 5, 15,
                     75, 96, 2, 8, 1, 1>
        <<<9960, 256, 0, stream>>>(total, w1h, g1, b1, h1, nullptr);

    // ---- conv2: h1 -> h2 ----  nTiles=9760, NT=4 -> 2440 blocks
    conv_wmma_kernel<_Float16, 32, 40, 498, 64, 20, 122, 2, 4, 2, 2, 5, 15,
                     2400, 2400, 4, 4, 1, 1>
        <<<2440, 256, 0, stream>>>(h1, w2h, g2, b2, h2, nullptr);

    // ---- conv3: h2 -> h3 ----  nTiles=1120, NT=2 -> 560 blocks
    conv_wmma_kernel<_Float16, 64, 20, 122, 128, 10, 28, 2, 4, 2, 2, 5, 15,
                     4800, 4800, 8, 2, 1, 1>
        <<<560, 256, 0, stream>>>(h2, w3h, g3, b3, h3, nullptr);

    // ---- conv4: h3 -> h4 ----  nTiles=100, NT=2, 2 M-blocks -> 100 blocks
    conv_wmma_kernel<_Float16, 128, 10, 28, 256, 5, 5, 2, 4, 2, 2, 5, 15,
                     9600, 9600, 8, 2, 2, 1>
        <<<100, 256, 0, stream>>>(h3, w4h, g4, b4, h4, nullptr);

    // ---- head: pool + FCs + top-2 ----
    head_kernel<<<64, 256, 0, stream>>>(h4, fw1, fb1, g1d, b1d, fw2, fb2, out_rw, sel);

    // ---- gather selected experts -> stem input (f16) ----
    gather_kernel<<<(4096000 + 255) / 256, 256, 0, stream>>>(x, sel, gt);

    // ---- stem conv: gt -> d_out (f32) ----  nTiles=63520, NT=4 -> 15880 blocks
    conv_wmma_kernel<_Float16, 1, 80, 800, 64, 40, 397, 2, 2, 3, 3, 7, 14,
                     98, 128, 4, 4, 1, 0>
        <<<15880, 256, 0, stream>>>(gt, wch, nullptr, nullptr, nullptr, out_conv);
}
